// MultiHeadedAttention_62843961475209
// MI455X (gfx1250) — compile-verified, matrix operand order checked by reading, not run on hardware
//
#include <hip/hip_runtime.h>
#include <math.h>

// MI455X / gfx1250, wave32. All heavy matmuls via v_wmma_f32_16x16x32_bf16.
// Dims fixed by the reference: B=4, T=4096, E=1024.

typedef __attribute__((ext_vector_type(16))) __bf16 v16bf;
typedef __attribute__((ext_vector_type(8)))  float  v8f;

static constexpr int Bn = 4;
static constexpr int Tn = 4096;
static constexpr int En = 1024;
static constexpr int BT = Bn * Tn;

// ---------------------------------------------------------------------------
// WMMA fragment helpers (CDNA5 wave32 layouts, CDNA5 ISA §7.12.2)
// ---------------------------------------------------------------------------

// A-matrix 16x32 bf16, row-major in LDS with stride 32 elements.
// Lane l (l<16): row M=l, K in {0..7, 16..23}; lane l+16: same row, K in {8..15, 24..31}.
// Two contiguous 16B runs per lane -> ds_load_b128 pairs.
__device__ inline v16bf load_frag_a(const __bf16* __restrict__ As) {
  const int l = threadIdx.x & 31;
  const __bf16* p = As + (l & 15) * 32 + ((l >> 4) << 3);
  v16bf f;
#pragma unroll
  for (int i = 0; i < 8; ++i) f[i] = p[i];
#pragma unroll
  for (int i = 0; i < 8; ++i) f[8 + i] = p[16 + i];
  return f;
}

// B tiles are staged into LDS pre-swizzled into fragment order:
// per 32x16 column tile, lane (kh*16 + n) owns 16 contiguous bf16.
__device__ inline v16bf load_frag_b(const __bf16* __restrict__ Bs, int ntile) {
  const int l = threadIdx.x & 31;
  const __bf16* p = Bs + ntile * 512 + ((l >> 4) * 16 + (l & 15)) * 16;
  v16bf f;
#pragma unroll
  for (int i = 0; i < 16; ++i) f[i] = p[i];
  return f;
}

// Swizzled LDS index for B element (k in 0..31, col c in 0..127).
__device__ inline int bswz(int k, int c) {
  const int nt = c >> 4, n = c & 15;
  const int kh = (k >> 3) & 1;            // lane half holding this K
  const int e  = (k & 7) + ((k >> 4) << 3); // element index within fragment
  return nt * 512 + (kh * 16 + n) * 16 + e;
}

__device__ inline v8f wmma_bf16(v16bf a, v16bf b, v8f c) {
  return __builtin_amdgcn_wmma_f32_16x16x32_bf16(false, a, false, b, (short)0, c,
                                                 false, false);
}

__device__ inline float block_reduce_sum(float v, float* sbuf) {
  const int tid = threadIdx.x;
#pragma unroll
  for (int off = 16; off > 0; off >>= 1) v += __shfl_xor(v, off, 32);
  if ((tid & 31) == 0) sbuf[tid >> 5] = v;
  __syncthreads();
  float r = 0.0f;
  if (tid < 8) {
    r = sbuf[tid];
#pragma unroll
    for (int off = 4; off > 0; off >>= 1) r += __shfl_xor(r, off, 32);
    if (tid == 0) sbuf[0] = r;
  }
  __syncthreads();
  r = sbuf[0];
  __syncthreads();
  return r;
}

// ---------------------------------------------------------------------------
// Small prep kernels
// ---------------------------------------------------------------------------

__global__ __launch_bounds__(256) void wv_to_bf16_kernel(
    const float* __restrict__ Wv, __bf16* __restrict__ out) {
  const int i = blockIdx.x * 256 + threadIdx.x;
  out[i] = (__bf16)Wv[i];
}

// Per token: q = token_row . Wq + bq ; k = token_row . Wk + bk. One wave/row.
__global__ __launch_bounds__(256) void qk_scalars_kernel(
    const float* __restrict__ token, const float* __restrict__ Wq,
    const float* __restrict__ bq, const float* __restrict__ Wk,
    const float* __restrict__ bk, float* __restrict__ qA, float* __restrict__ kA) {
  const int wave = threadIdx.x >> 5, lane = threadIdx.x & 31;
  const int row = blockIdx.x * 8 + wave;
  const float* tr = token + (size_t)row * En;
  float sq = 0.f, sk = 0.f;
#pragma unroll 4
  for (int j = lane; j < En; j += 32) {
    const float t = tr[j];
    sq += t * Wq[j];
    sk += t * Wk[j];
  }
#pragma unroll
  for (int off = 16; off > 0; off >>= 1) {
    sq += __shfl_xor(sq, off, 32);
    sk += __shfl_xor(sk, off, 32);
  }
  if (lane == 0) { qA[row] = sq + bq[0]; kA[row] = sk + bk[0]; }
}

// Per batch: max/min of k over T (gives the analytic softmax max).
__global__ __launch_bounds__(256) void kstats_kernel(
    const float* __restrict__ kA, float* __restrict__ kmx, float* __restrict__ kmn) {
  __shared__ float smax[8], smin[8];
  const int b = blockIdx.x, tid = threadIdx.x;
  const float* kb = kA + ((size_t)b << 12);
  float mx = -3.0e38f, mn = 3.0e38f;
  for (int x = tid; x < Tn; x += 256) {
    const float v = kb[x];
    mx = fmaxf(mx, v);
    mn = fminf(mn, v);
  }
#pragma unroll
  for (int off = 16; off > 0; off >>= 1) {
    mx = fmaxf(mx, __shfl_xor(mx, off, 32));
    mn = fminf(mn, __shfl_xor(mn, off, 32));
  }
  if ((tid & 31) == 0) { smax[tid >> 5] = mx; smin[tid >> 5] = mn; }
  __syncthreads();
  if (tid < 8) {
    mx = smax[tid]; mn = smin[tid];
#pragma unroll
    for (int off = 4; off > 0; off >>= 1) {
      mx = fmaxf(mx, __shfl_xor(mx, off, 32));
      mn = fminf(mn, __shfl_xor(mn, off, 32));
    }
    if (tid == 0) { kmx[b] = mx; kmn[b] = mn; }
  }
}

// Per token: reciprocal softmax denominator. One wave/row.
__global__ __launch_bounds__(256) void lsum_kernel(
    const float* __restrict__ qA, const float* __restrict__ kA,
    const float* __restrict__ kmx, const float* __restrict__ kmn,
    float* __restrict__ rlA) {
  const int wave = threadIdx.x >> 5, lane = threadIdx.x & 31;
  const int row = blockIdx.x * 8 + wave;
  const int b = row >> 12;
  const float qv = qA[row];
  const float m = qv * (qv >= 0.f ? kmx[b] : kmn[b]);
  const float* kb = kA + ((size_t)b << 12);
  float s = 0.f;
#pragma unroll 4
  for (int x = lane; x < Tn; x += 32) s += __expf(qv * kb[x] - m);
#pragma unroll
  for (int off = 16; off > 0; off >>= 1) s += __shfl_xor(s, off, 32);
  if (lane == 0) rlA[row] = 1.0f / s;
}

// ---------------------------------------------------------------------------
// GEMM 1: value = token @ Wv + bv  (M=16384, N=1024, K=1024), out bf16
// Block = 256 thr (8 waves), block tile 128x128, wave = 16 rows x 128 cols.
// ---------------------------------------------------------------------------
__global__ __launch_bounds__(256) void value_gemm_kernel(
    const float* __restrict__ token, const __bf16* __restrict__ Wvbf,
    const float* __restrict__ bv, __bf16* __restrict__ vbf) {
  __shared__ __bf16 As[128 * 32];
  __shared__ __bf16 Bs[8 * 512];
  const int tid = threadIdx.x, wave = tid >> 5;
  const int mBase = blockIdx.y * 128;
  const int nBase = blockIdx.x * 128;
  v8f acc[8] = {};
  for (int kt = 0; kt < En / 32; ++kt) {
    const int k0 = kt * 32;
#pragma unroll
    for (int i = 0; i < 16; ++i) {  // stage A tile (f32 -> bf16)
      const int idx = i * 256 + tid;
      const int r = idx >> 5, c = idx & 31;
      As[r * 32 + c] = (__bf16)token[(size_t)(mBase + r) * En + k0 + c];
    }
#pragma unroll
    for (int i = 0; i < 16; ++i) {  // stage B tile, swizzled to fragment order
      const int idx = i * 256 + tid;
      const int r = idx >> 7, c = idx & 127;
      Bs[bswz(r, c)] = Wvbf[(size_t)(k0 + r) * En + nBase + c];
    }
    if (kt + 1 < En / 32)  // hint: pull next A strip toward the WGP
      __builtin_prefetch(&token[(size_t)(mBase + (tid >> 1)) * En + k0 + 32], 0, 0);
    __syncthreads();
    const v16bf a = load_frag_a(As + wave * 16 * 32);
#pragma unroll
    for (int nt = 0; nt < 8; ++nt)
      acc[nt] = wmma_bf16(a, load_frag_b(Bs, nt), acc[nt]);
    __syncthreads();
  }
  const int lane = tid & 31, n = lane & 15, mh = (lane >> 4) << 3;
  const int rowBase = mBase + wave * 16 + mh;
#pragma unroll
  for (int nt = 0; nt < 8; ++nt) {
    const int col = nBase + nt * 16 + n;
    const float bias = bv[col];
#pragma unroll
    for (int i = 0; i < 8; ++i)
      vbf[(size_t)(rowBase + i) * En + col] = (__bf16)(acc[nt][i] + bias);
  }
}

// ---------------------------------------------------------------------------
// Attention: z[b,t,e] = sum_x softmax_x(q_t k_x) * value[b,x,e]
// Flash-style: P fragment synthesized in registers (rank-1 scores), never
// materialized. Per batch: M=T (t), N=E (e), K=T (x).
// ---------------------------------------------------------------------------
__global__ __launch_bounds__(256) void attn_kernel(
    const __bf16* __restrict__ vbf, const float* __restrict__ qA,
    const float* __restrict__ kA, const float* __restrict__ rlA,
    const float* __restrict__ kmx, const float* __restrict__ kmn,
    float* __restrict__ z) {
  __shared__ __bf16 Vs[8 * 512];
  __shared__ float ks[32];
  const int tid = threadIdx.x, wave = tid >> 5, lane = tid & 31;
  const int b = blockIdx.z;
  const int tBase = blockIdx.y * 128;
  const int eBase = blockIdx.x * 128;
  const int t = tBase + wave * 16 + (lane & 15);
  const size_t rowIdx = ((size_t)b << 12) + t;
  const float qv = qA[rowIdx];
  const float rl = rlA[rowIdx];
  const float m = qv * (qv >= 0.f ? kmx[b] : kmn[b]);
  const int kh8 = ((lane >> 4) & 1) << 3;
  const __bf16* vb = vbf + (((size_t)b << 12) * En);
  const float* kb = kA + ((size_t)b << 12);
  v8f acc[8] = {};
  for (int xt = 0; xt < Tn / 32; ++xt) {
    const int x0 = xt * 32;
#pragma unroll
    for (int i = 0; i < 16; ++i) {  // stage V tile (bf16), swizzled
      const int idx = i * 256 + tid;
      const int r = idx >> 7, c = idx & 127;
      Vs[bswz(r, c)] = vb[(size_t)(x0 + r) * En + eBase + c];
    }
    if (tid < 32) ks[tid] = kb[x0 + tid];
    __syncthreads();
    // Build normalized-probability A fragment in registers: 16 exps/lane.
    v16bf a;
#pragma unroll
    for (int i = 0; i < 8; ++i)
      a[i] = (__bf16)(__expf(qv * ks[kh8 + i] - m) * rl);
#pragma unroll
    for (int i = 0; i < 8; ++i)
      a[8 + i] = (__bf16)(__expf(qv * ks[16 + kh8 + i] - m) * rl);
#pragma unroll
    for (int nt = 0; nt < 8; ++nt)
      acc[nt] = wmma_bf16(a, load_frag_b(Vs, nt), acc[nt]);
    __syncthreads();
  }
  const int n = lane & 15, mh = (lane >> 4) << 3;
  const size_t rowBase = ((size_t)b << 12) + tBase + wave * 16 + mh;
#pragma unroll
  for (int nt = 0; nt < 8; ++nt) {
    const int col = eBase + nt * 16 + n;
#pragma unroll
    for (int i = 0; i < 8; ++i)
      z[(rowBase + i) * En + col] = acc[nt][i];
  }
}

// ---------------------------------------------------------------------------
// LayerNorm 1: z -> normalized, written as bf16 (A + residual for GEMM 2)
// ---------------------------------------------------------------------------
__global__ __launch_bounds__(256) void ln1_kernel(
    const float* __restrict__ z, const float* __restrict__ gamma,
    const float* __restrict__ beta, __bf16* __restrict__ zlnbf) {
  __shared__ float sbuf[8];
  const size_t row = blockIdx.x;
  const float* zr = z + row * En;
  const int tid = threadIdx.x;
  float v[4];
  float s = 0.f;
#pragma unroll
  for (int i = 0; i < 4; ++i) { v[i] = zr[tid + i * 256]; s += v[i]; }
  s = block_reduce_sum(s, sbuf);
  const float mu = s * (1.0f / 1024.0f);
  float s2 = 0.f;
#pragma unroll
  for (int i = 0; i < 4; ++i) { const float d = v[i] - mu; s2 += d * d; }
  s2 = block_reduce_sum(s2, sbuf);
  const float inv = rsqrtf(s2 * (1.0f / 1024.0f) + 1e-3f);
#pragma unroll
  for (int i = 0; i < 4; ++i) {
    const int c = tid + i * 256;
    zlnbf[row * En + c] = (__bf16)((v[i] - mu) * inv * gamma[c] + beta[c]);
  }
}

// ---------------------------------------------------------------------------
// GEMM 2: z2 = zln @ Wv + bv + zln  (residual), f32 out
// ---------------------------------------------------------------------------
__global__ __launch_bounds__(256) void gemm2_kernel(
    const __bf16* __restrict__ zlnbf, const __bf16* __restrict__ Wvbf,
    const float* __restrict__ bv, float* __restrict__ z2) {
  __shared__ __bf16 As[128 * 32];
  __shared__ __bf16 Bs[8 * 512];
  const int tid = threadIdx.x, wave = tid >> 5;
  const int mBase = blockIdx.y * 128;
  const int nBase = blockIdx.x * 128;
  v8f acc[8] = {};
  for (int kt = 0; kt < En / 32; ++kt) {
    const int k0 = kt * 32;
#pragma unroll
    for (int i = 0; i < 16; ++i) {
      const int idx = i * 256 + tid;
      const int r = idx >> 5, c = idx & 31;
      As[r * 32 + c] = zlnbf[(size_t)(mBase + r) * En + k0 + c];
    }
#pragma unroll
    for (int i = 0; i < 16; ++i) {
      const int idx = i * 256 + tid;
      const int r = idx >> 7, c = idx & 127;
      Bs[bswz(r, c)] = Wvbf[(size_t)(k0 + r) * En + nBase + c];
    }
    __syncthreads();
    const v16bf a = load_frag_a(As + wave * 16 * 32);
#pragma unroll
    for (int nt = 0; nt < 8; ++nt)
      acc[nt] = wmma_bf16(a, load_frag_b(Bs, nt), acc[nt]);
    __syncthreads();
  }
  const int lane = tid & 31, n = lane & 15, mh = (lane >> 4) << 3;
  const int rowBase = mBase + wave * 16 + mh;
#pragma unroll
  for (int nt = 0; nt < 8; ++nt) {
    const int col = nBase + nt * 16 + n;
    const float bias = bv[col];
#pragma unroll
    for (int i = 0; i < 8; ++i) {
      const float res = (float)zlnbf[(size_t)(rowBase + i) * En + col];
      z2[(size_t)(rowBase + i) * En + col] = acc[nt][i] + bias + res;
    }
  }
}

// ---------------------------------------------------------------------------
// LayerNorm 2 + ReLU -> final f32 output
// ---------------------------------------------------------------------------
__global__ __launch_bounds__(256) void ln2_relu_kernel(
    const float* __restrict__ z2, const float* __restrict__ gamma,
    const float* __restrict__ beta, float* __restrict__ out) {
  __shared__ float sbuf[8];
  const size_t row = blockIdx.x;
  const float* zr = z2 + row * En;
  const int tid = threadIdx.x;
  float v[4];
  float s = 0.f;
#pragma unroll
  for (int i = 0; i < 4; ++i) { v[i] = zr[tid + i * 256]; s += v[i]; }
  s = block_reduce_sum(s, sbuf);
  const float mu = s * (1.0f / 1024.0f);
  float s2 = 0.f;
#pragma unroll
  for (int i = 0; i < 4; ++i) { const float d = v[i] - mu; s2 += d * d; }
  s2 = block_reduce_sum(s2, sbuf);
  const float inv = rsqrtf(s2 * (1.0f / 1024.0f) + 1e-3f);
#pragma unroll
  for (int i = 0; i < 4; ++i) {
    const int c = tid + i * 256;
    out[row * En + c] = fmaxf(0.0f, (v[i] - mu) * inv * gamma[c] + beta[c]);
  }
}

// ---------------------------------------------------------------------------
// Launch
// ---------------------------------------------------------------------------
extern "C" void kernel_launch(void* const* d_in, const int* in_sizes, int n_in,
                              void* d_out, int out_size, void* d_ws, size_t ws_size,
                              hipStream_t stream) {
  (void)in_sizes; (void)n_in; (void)out_size; (void)ws_size;
  const float* token  = (const float*)d_in[0];
  const float* Wk     = (const float*)d_in[1];
  const float* bk     = (const float*)d_in[2];
  const float* Wq     = (const float*)d_in[3];
  const float* bq     = (const float*)d_in[4];
  const float* Wv     = (const float*)d_in[5];
  const float* bv     = (const float*)d_in[6];
  const float* gamma1 = (const float*)d_in[7];
  const float* beta1  = (const float*)d_in[8];
  const float* gamma2 = (const float*)d_in[9];
  const float* beta2  = (const float*)d_in[10];
  float* out = (float*)d_out;

  // Workspace layout (bytes). vbf region is reused for zln bf16; z region for z2.
  char* ws = (char*)d_ws;
  float*  qA   = (float*)(ws + 0);                                   // 64 KB
  float*  kA   = (float*)(ws + (64 << 10));                          // 64 KB
  float*  rlA  = (float*)(ws + (128 << 10));                         // 64 KB
  float*  kmx  = (float*)(ws + (192 << 10));                         // 512 B
  float*  kmn  = (float*)(ws + (192 << 10) + 512);                   // 512 B
  __bf16* wvbf = (__bf16*)(ws + (256 << 10));                        // 2 MB
  __bf16* vbf  = (__bf16*)(ws + (256 << 10) + (2 << 20));            // 32 MB (reused: zln bf16)
  float*  zbuf = (float*)(ws + (256 << 10) + (2 << 20) + (32 << 20)); // 64 MB (reused: z2)

  wv_to_bf16_kernel<<<(En * En) / 256, 256, 0, stream>>>(Wv, wvbf);
  qk_scalars_kernel<<<BT / 8, 256, 0, stream>>>(token, Wq, bq, Wk, bk, qA, kA);
  kstats_kernel<<<Bn, 256, 0, stream>>>(kA, kmx, kmn);
  lsum_kernel<<<BT / 8, 256, 0, stream>>>(qA, kA, kmx, kmn, rlA);
  value_gemm_kernel<<<dim3(En / 128, BT / 128), 256, 0, stream>>>(token, wvbf, bv, vbf);
  attn_kernel<<<dim3(En / 128, Tn / 128, Bn), 256, 0, stream>>>(vbf, qA, kA, rlA, kmx,
                                                                kmn, zbuf);
  ln1_kernel<<<BT, 256, 0, stream>>>(zbuf, gamma1, beta1, vbf /* -> zln bf16 */);
  gemm2_kernel<<<dim3(En / 128, BT / 128), 256, 0, stream>>>(vbf, wvbf, bv,
                                                             zbuf /* -> z2 */);
  ln2_relu_kernel<<<BT, 256, 0, stream>>>(zbuf, gamma2, beta2, out);
}
